// ScorerLayer_35957466202775
// MI455X (gfx1250) — matrix operand order repaired and checked by reading, not compiled, not run on hardware
//
#include <hip/hip_runtime.h>

// ---------------------------------------------------------------------------
// ScorerLayer on MI455X (gfx1250), bf16 WMMA with f32 accumulate.
// B=2, N=2048, T=32, D=256, H=1024.
// feats@W1 = (p*r)@Wa + p@(Wb+Wc) + r@(Wd-Wb);  out = relu(.)@W2 + b2
// ---------------------------------------------------------------------------

typedef __attribute__((ext_vector_type(16))) __bf16 v16bf;
typedef __attribute__((ext_vector_type(8)))  float  v8f;

#define B_  2
#define N_  2048
#define T_  32
#define D_  256
#define H_  1024

#define PROW 260        // padded f32 row length for p/r staging (bank-conflict free)
#define BCOL 264        // padded bf16 column length for B slab staging

#define P_OFF   0
#define R_OFF   (16 * PROW * 4)            // 16640
#define PQ_OFF  (R_OFF + 32 * PROW * 4)    // 49920
#define RQ_OFF  (PQ_OFF + 16 * H_ * 4)     // 115456
#define BS_OFF  (RQ_OFF + 32 * H_ * 4)     // 246528
#define LDS_BYTES (BS_OFF + 64 * BCOL * 2) // 280320 bytes (< 320KB WGP LDS)

// ---------------------------------------------------------------------------
// Prep: build transposed bf16 weight blocks in workspace.
//   Wa_t[col][k] = bf16(W1[k][col])                (k in 0..255, p*r block)
//   Wp_t[col][k] = bf16(W1[256+k][col] + W1[512+k][col])   (p block)
//   Wr_t[col][k] = bf16(W1[768+k][col] - W1[256+k][col])   (r block)
// ---------------------------------------------------------------------------
__global__ void prep_weights(const float* __restrict__ W1,
                             unsigned short* __restrict__ Wa,
                             unsigned short* __restrict__ Wp,
                             unsigned short* __restrict__ Wr) {
  int tid = blockIdx.x * blockDim.x + threadIdx.x;  // 0 .. 262143
  int col = tid & (H_ - 1);
  int k   = tid >> 10;                              // 0 .. 255
  float a = W1[(size_t)k * H_ + col];
  float b = W1[(size_t)(256 + k) * H_ + col];
  float c = W1[(size_t)(512 + k) * H_ + col];
  float d = W1[(size_t)(768 + k) * H_ + col];
  union { __bf16 h; unsigned short u; } ua, up, ur;
  ua.h = (__bf16)a;
  up.h = (__bf16)(b + c);
  ur.h = (__bf16)(d - b);
  size_t o = (size_t)col * 256 + k;
  Wa[o] = ua.u;
  Wp[o] = up.u;
  Wr[o] = ur.u;
}

// ---------------------------------------------------------------------------
// Fragment helpers (A/B: this code's contiguous k order; C layout per CDNA5 ISA 7.12.2):
//  A 16-bit 16x32: lane l holds row M=l%16; VGPR0-3: K = 16*(l/16)+0..7,
//                  VGPR4-7: K = 16*(l/16)+8..15   (within 32-k window)
//  B 16-bit 32x16: lane l holds col N=l%16; K = 16*(l/16)+0..15 contiguous
//  C f32 16x16:    VGPR v: lanes 0-15 -> (M=v, N=l), lanes 16-31 -> (M=v+8)
// ---------------------------------------------------------------------------
__device__ __forceinline__ v16bf load_bfrag(const unsigned short* base, int khalf) {
  union { v16bf v; float4 q[2]; } u;
  const float4* s = (const float4*)(base + khalf * 16);
  u.q[0] = s[0];
  u.q[1] = s[1];
  return u.v;
}

__device__ __forceinline__ v16bf make_frag_mul(const float* pr, const float* rr,
                                               int f, int khalf) {
  const int k1 = f * 32 + khalf * 16;
  const int k2 = k1 + 8;
  float pv[16], rv[16];
  *(float4*)(pv + 0)  = *(const float4*)(pr + k1);
  *(float4*)(pv + 4)  = *(const float4*)(pr + k1 + 4);
  *(float4*)(pv + 8)  = *(const float4*)(pr + k2);
  *(float4*)(pv + 12) = *(const float4*)(pr + k2 + 4);
  *(float4*)(rv + 0)  = *(const float4*)(rr + k1);
  *(float4*)(rv + 4)  = *(const float4*)(rr + k1 + 4);
  *(float4*)(rv + 8)  = *(const float4*)(rr + k2);
  *(float4*)(rv + 12) = *(const float4*)(rr + k2 + 4);
  union { v16bf v; __bf16 e[16]; } u;
#pragma unroll
  for (int i = 0; i < 16; ++i) u.e[i] = (__bf16)(pv[i] * rv[i]);
  return u.v;
}

__device__ __forceinline__ v16bf make_frag(const float* pr, int f, int khalf) {
  const int k1 = f * 32 + khalf * 16;
  const int k2 = k1 + 8;
  float pv[16];
  *(float4*)(pv + 0)  = *(const float4*)(pr + k1);
  *(float4*)(pv + 4)  = *(const float4*)(pr + k1 + 4);
  *(float4*)(pv + 8)  = *(const float4*)(pr + k2);
  *(float4*)(pv + 12) = *(const float4*)(pr + k2 + 4);
  union { v16bf v; __bf16 e[16]; } u;
#pragma unroll
  for (int i = 0; i < 16; ++i) u.e[i] = (__bf16)pv[i];
  return u.v;
}

__device__ __forceinline__ v8f wmma_bf16(v16bf a, v16bf b, v8f c) {
  return __builtin_amdgcn_wmma_f32_16x16x32_bf16(false, a, false, b, (short)0, c,
                                                 false, false);
}

// ---------------------------------------------------------------------------
// Main fused kernel. Grid: (N/16, B). Block: 512 threads (16 waves).
// Row mapping within a block: row = n_local*32 + t  (512 rows = 32 M-tiles,
// each M-tile shares one n_local; wave w owns M-tiles w and w+16).
// ---------------------------------------------------------------------------
__launch_bounds__(512, 1)
__global__ void scorer_main(const float* __restrict__ P,
                            const float* __restrict__ R,
                            const unsigned short* __restrict__ Wa,
                            const unsigned short* __restrict__ Wp,
                            const unsigned short* __restrict__ Wr,
                            const float* __restrict__ b1,
                            const float* __restrict__ W2,
                            const float* __restrict__ b2,
                            float* __restrict__ out) {
  extern __shared__ char smem[];
  float* p_lds  = (float*)(smem + P_OFF);           // 16 x PROW f32
  float* r_lds  = (float*)(smem + R_OFF);           // 32 x PROW f32
  float* pq_lds = (float*)(smem + PQ_OFF);          // 16 x 1024 f32
  float* rq_lds = (float*)(smem + RQ_OFF);          // 32 x 1024 f32
  unsigned short* b_lds = (unsigned short*)(smem + BS_OFF);  // 64 x BCOL bf16

  const int tid   = threadIdx.x;
  const int wave  = tid >> 5;    // 0..15
  const int lane  = tid & 31;
  const int l16   = lane & 15;
  const int khalf = lane >> 4;   // 0 or 1
  const int bidx  = blockIdx.y;
  const int n0    = blockIdx.x << 4;

  // ---- phase 0: stage p (16x256) and r (32x256) into padded LDS ----
  {
    int row = tid >> 5;                 // 16 rows, 32 threads each
    int k0  = (tid & 31) << 3;          // 8 floats
    const float4* s = (const float4*)(P + ((size_t)bidx * N_ + n0 + row) * D_ + k0);
    float4* dst = (float4*)(p_lds + row * PROW + k0);
    dst[0] = s[0];
    dst[1] = s[1];
  }
  {
    int row = tid >> 4;                 // 32 rows, 16 threads each
    int k0  = (tid & 15) << 4;          // 16 floats
    const float4* s = (const float4*)(R + ((size_t)bidx * T_ + row) * D_ + k0);
    float4* dst = (float4*)(r_lds + row * PROW + k0);
    dst[0] = s[0]; dst[1] = s[1]; dst[2] = s[2]; dst[3] = s[3];
  }
  __syncthreads();

  // ---- phase 1: PQ = p @ (Wb+Wc)  -> pq_lds[16][1024] ----
  {
    v16bf apq[8];
    const float* prow = p_lds + l16 * PROW;
#pragma unroll
    for (int f = 0; f < 8; ++f) apq[f] = make_frag(prow, f, khalf);
#pragma unroll
    for (int nt = 0; nt < 4; ++nt) {
      const int col = (wave * 4 + nt) * 16 + l16;
      v8f c = {};
#pragma unroll
      for (int f = 0; f < 8; ++f)
        c = wmma_bf16(apq[f], load_bfrag(Wp + (size_t)col * 256 + f * 32, khalf), c);
#pragma unroll
      for (int v = 0; v < 8; ++v)
        pq_lds[(v + 8 * khalf) * H_ + col] = c[v];
    }
  }

  // ---- phase 2: RQ = r @ (Wd-Wb) + b1 -> rq_lds[32][1024] ----
#pragma unroll
  for (int mt = 0; mt < 2; ++mt) {
    v16bf arq[8];
    const float* rrow = r_lds + (mt * 16 + l16) * PROW;
#pragma unroll
    for (int f = 0; f < 8; ++f) arq[f] = make_frag(rrow, f, khalf);
#pragma unroll
    for (int nt = 0; nt < 4; ++nt) {
      const int col = (wave * 4 + nt) * 16 + l16;
      v8f c = {};
#pragma unroll
      for (int f = 0; f < 8; ++f)
        c = wmma_bf16(arq[f], load_bfrag(Wr + (size_t)col * 256 + f * 32, khalf), c);
      const float b1v = b1[col];
#pragma unroll
      for (int v = 0; v < 8; ++v)
        rq_lds[(mt * 16 + v + 8 * khalf) * H_ + col] = c[v] + b1v;
    }
  }

  // ---- phase 3: build bf16(p*r) A-fragments for this wave's 2 M-tiles ----
  v16bf amain[2][8];
  int nlocal[2], tbase[2];
#pragma unroll
  for (int mi = 0; mi < 2; ++mi) {
    const int m = wave + 16 * mi;            // M-tile 0..31
    nlocal[mi]  = m >> 1;
    tbase[mi]   = (m & 1) * 16 + 8 * khalf;  // t of C-row v is tbase+v
    const float* prow = p_lds + nlocal[mi] * PROW;
    const float* rrow = r_lds + ((m & 1) * 16 + l16) * PROW;
#pragma unroll
    for (int f = 0; f < 8; ++f) amain[mi][f] = make_frag_mul(prow, rrow, f, khalf);
  }

  // ---- phase 4: slab loop over Wa columns; fused relu + W2 reduction ----
  float acc[2][8] = {};
  for (int s = 0; s < 16; ++s) {
    __syncthreads();   // previous slab consumed / phase1-2 LDS visible
    {
      int cl = tid >> 3;                 // 64 local cols, 8 threads each
      int k0 = (tid & 7) << 5;           // 32 bf16 = 64B
      const float4* src = (const float4*)(Wa + (size_t)(s * 64 + cl) * 256 + k0);
      float4* dst = (float4*)(b_lds + cl * BCOL + k0);
      dst[0] = src[0]; dst[1] = src[1]; dst[2] = src[2]; dst[3] = src[3];
    }
    __syncthreads();

#pragma unroll
    for (int mi = 0; mi < 2; ++mi) {
#pragma unroll
      for (int nt = 0; nt < 4; ++nt) {
        const int cl = nt * 16 + l16;    // local column in slab
        v8f c = {};
#pragma unroll
        for (int f = 0; f < 8; ++f) {
          union { v16bf v; float4 q[2]; } u;
          const float4* bs = (const float4*)(b_lds + cl * BCOL + f * 32 + khalf * 16);
          u.q[0] = bs[0];
          u.q[1] = bs[1];
          c = wmma_bf16(amain[mi][f], u.v, c);
        }
        const int col   = s * 64 + cl;
        const float w2v = W2[col];
        const float pqv = pq_lds[nlocal[mi] * H_ + col];
#pragma unroll
        for (int v = 0; v < 8; ++v) {
          float x = c[v] + pqv + rq_lds[(tbase[mi] + v) * H_ + col];
          x = fmaxf(x, 0.0f);
          acc[mi][v] = fmaf(x, w2v, acc[mi][v]);
        }
      }
    }
  }

  // ---- phase 5: butterfly-reduce the 16 column classes; store scores ----
  const float b2v = b2[0];
#pragma unroll
  for (int mi = 0; mi < 2; ++mi) {
#pragma unroll
    for (int v = 0; v < 8; ++v) {
      float x = acc[mi][v];
      x += __shfl_xor(x, 1);
      x += __shfl_xor(x, 2);
      x += __shfl_xor(x, 4);
      x += __shfl_xor(x, 8);
      acc[mi][v] = x;
    }
    if (l16 == 0) {
      const int m = wave + 16 * mi;
      const int n = n0 + (m >> 1);
      const int t0 = (m & 1) * 16 + 8 * khalf;
      float* o = out + ((size_t)bidx * N_ + n) * T_ + t0;
#pragma unroll
      for (int v = 0; v < 8; ++v) o[v] = acc[mi][v] + b2v;
    }
  }
}

// ---------------------------------------------------------------------------
extern "C" void kernel_launch(void* const* d_in, const int* in_sizes, int n_in,
                              void* d_out, int out_size, void* d_ws, size_t ws_size,
                              hipStream_t stream) {
  const float* P  = (const float*)d_in[0];  // (B,N,D)
  const float* R  = (const float*)d_in[1];  // (B,T,D)
  const float* W1 = (const float*)d_in[2];  // (H,H)
  const float* b1 = (const float*)d_in[3];  // (H,)
  const float* W2 = (const float*)d_in[4];  // (H,1)
  const float* b2 = (const float*)d_in[5];  // (1,)
  float* out = (float*)d_out;               // (B,N,T)

  unsigned short* Wa = (unsigned short*)d_ws;        // 1024x256 bf16
  unsigned short* Wp = Wa + (size_t)H_ * 256;        // 1024x256 bf16
  unsigned short* Wr = Wp + (size_t)H_ * 256;        // 1024x256 bf16

  prep_weights<<<(H_ * 256) / 256, 256, 0, stream>>>(W1, Wa, Wp, Wr);

  dim3 grid(N_ / 16, B_);
  scorer_main<<<grid, 512, LDS_BYTES, stream>>>(P, R, Wa, Wp, Wr, b1, W2, b2, out);
}
